// DentalMetricDGCNN_49804440765100
// MI455X (gfx1250) — matrix operand order repaired and checked by reading, not compiled, not run on hardware
//
#include <hip/hip_runtime.h>

typedef __attribute__((ext_vector_type(16))) _Float16 v16h;
typedef __attribute__((ext_vector_type(8)))  float    v8f;

#define LN_EPS   1e-5f
#define KNBR     20
#define ARC_S    30.0f
#define ARC_M    0.4f

// ---------------------------------------------------------------- WMMA utils
__device__ static inline v8f wmma16(v16h a, v16h b, v8f c) {
  // D = A(16x32 f16) * B(32x16 f16) + C(16x16 f32)
  return __builtin_amdgcn_wmma_f32_16x16x32_f16(false, a, false, b, (short)0, c,
                                                false, false);
}

// A-matrix 16x32 f16 fragment from a row-major tile (lda in halves).
// Lane l -> M=l%16 ; element s holds K=k0+16*(l/16)+s (contiguous order, same as the B loaders).
__device__ static inline v16h load_a_frag(const _Float16* As, int lda, int k0,
                                          int lane) {
  const int m  = lane & 15;
  const int hi = lane >> 4;
  v16h a;
#pragma unroll
  for (int v = 0; v < 8; ++v) {
    const int kb = hi * 16 + 2 * v;
    const _Float16* p = As + m * lda + k0 + kb;
    a[2 * v]     = p[0];
    a[2 * v + 1] = p[1];
  }
  return a;
}

// B-matrix 32x16 f16 fragment from row-major global weights W[K][nout].
__device__ static inline v16h load_b_frag(const _Float16* W, int nout, int c0,
                                          int k0, int lane) {
  const int col = c0 + (lane & 15);
  const int kb  = k0 + (lane >> 4) * 16;
  v16h b;
#pragma unroll
  for (int e = 0; e < 16; ++e) b[e] = W[(size_t)(kb + e) * nout + col];
  return b;
}

// B-matrix fragment for a TRANSPOSED operand: B[k][n] = X[jbase+n][k], X
// row-major [*,64] f16. Per lane this is 16 contiguous halves -> b128 loads.
__device__ static inline v16h load_bt_frag(const _Float16* X, int jbase, int k0,
                                           int lane) {
  const int col = jbase + (lane & 15);
  const int kb  = k0 + (lane >> 4) * 16;
  const _Float16* p = X + (size_t)col * 64 + kb;
  v16h b;
#pragma unroll
  for (int e = 0; e < 16; ++e) b[e] = p[e];
  return b;
}

// LayerNorm over one row of length C (optionally ReLU), src/dst may alias.
template <typename T>
__device__ static inline void ln_row(const float* src, T* dst, const float* gg,
                                     const float* bb, int C, bool relu) {
  float m = 0.f;
  for (int c = 0; c < C; ++c) m += src[c];
  m /= (float)C;
  float v = 0.f;
  for (int c = 0; c < C; ++c) { float d = src[c] - m; v += d * d; }
  v /= (float)C;
  const float rinv = rsqrtf(v + LN_EPS);
  for (int c = 0; c < C; ++c) {
    float t = gg[c] * (src[c] - m) * rinv + bb[c];
    if (relu) t = fmaxf(t, 0.f);
    dst[c] = (T)t;
  }
}

// -------------------------------------------------- f32 -> f16 copy (+pad K)
__global__ void __launch_bounds__(256)
wconv_kernel(const float* __restrict__ src, _Float16* __restrict__ dst, int K,
             int Nn, int Kpad) {
  int t = blockIdx.x * 256 + threadIdx.x;
  if (t >= Kpad * Nn) return;
  int k = t / Nn, n = t - k * Nn;
  dst[t] = (_Float16)((k < K) ? src[(size_t)k * Nn + n] : 0.f);
}

// ------------------------------------------------------------- squared norms
__global__ void __launch_bounds__(256)
sqnorm_kernel(const float* __restrict__ xin, float* __restrict__ sq, int BN) {
  const int r = blockIdx.x * 256 + threadIdx.x;
  if (r >= BN) return;
  float s = 0.f;
#pragma unroll 8
  for (int c = 0; c < 64; ++c) {
    const float v = xin[(size_t)r * 64 + c];
    s += v * v;
  }
  sq[r] = s;
}

// ------------------------------------------- scalar kNN (layer 1 only, F=6)
template <int F>
__global__ void __launch_bounds__(256)
knn_kernel(const float* __restrict__ xin, int* __restrict__ oidx, int N) {
  __shared__ float tile[128 * F];
  const int nb = N >> 8;
  const int b  = blockIdx.x / nb;
  const int i  = (blockIdx.x % nb) * 256 + threadIdx.x;

  float xi[F];
#pragma unroll
  for (int f = 0; f < F; ++f) xi[f] = xin[((size_t)b * N + i) * F + f];

  float topd[KNBR];
  int   topi[KNBR];
#pragma unroll
  for (int q = 0; q < KNBR; ++q) { topd[q] = 3.0e38f; topi[q] = 0; }
  float worst = 3.0e38f;
  int   wpos  = 0;

  for (int j0 = 0; j0 < N; j0 += 128) {
    __syncthreads();
    for (int t = threadIdx.x; t < 128 * F; t += 256)
      tile[t] = xin[((size_t)b * N + j0 + t / F) * F + (t % F)];
    __syncthreads();
    for (int j = 0; j < 128; ++j) {
      float d = 0.f;
#pragma unroll
      for (int f = 0; f < F; ++f) {
        float df = xi[f] - tile[j * F + f];
        d += df * df;
      }
      if (d < worst) {
        const int jj = j0 + j;
#pragma unroll
        for (int q = 0; q < KNBR; ++q)
          if (q == wpos) { topd[q] = d; topi[q] = jj; }
        worst = -3.0e38f;
#pragma unroll
        for (int q = 0; q < KNBR; ++q)
          if (topd[q] > worst) { worst = topd[q]; wpos = q; }
      }
    }
  }
#pragma unroll
  for (int q = 0; q < KNBR; ++q)
    oidx[((size_t)b * N + i) * KNBR + q] = topi[q];
}

// ------------------------- WMMA Gram-matrix kNN (layers 2/3, F=64, f16 dots)
// One wave handles 16 query rows. Rank key = ||x_j||^2 - 2*<x_i,x_j>
// (||x_i||^2 is row-constant, drops out of top-K). Per iteration two 16x16
// dot tiles via 4 WMMAs; lanes 0-15 scan even tiles, lanes 16-31 odd tiles;
// the two partial top-20 lists per row are merged at the end via LDS.
__global__ void __launch_bounds__(32)
knn_wmma_kernel(const _Float16* __restrict__ xh, const float* __restrict__ sq,
                int* __restrict__ oidx, int N) {
  __shared__ float Ds[2 * 256];        // two 16x16 dot tiles
  __shared__ float Mrg_d[16 * 2 * KNBR];
  __shared__ int   Mrg_i[16 * 2 * KNBR];

  const int lane = threadIdx.x;
  const int nb   = N >> 4;
  const int b    = blockIdx.x / nb;
  const int i0   = (blockIdx.x % nb) << 4;
  const _Float16* xb  = xh + (size_t)b * N * 64;
  const float*    sqb = sq + (size_t)b * N;

  // A fragments for rows i0..i0+15 over K=64 (two 32-chunks), straight from
  // global f16 activations (row-major, lda=64).
  const v16h a0 = load_a_frag(xb + (size_t)i0 * 64, 64, 0, lane);
  const v16h a1 = load_a_frag(xb + (size_t)i0 * 64, 64, 32, lane);

  const int row  = lane & 15;
  const int tsel = lane >> 4;  // which of the two tiles this lane scans

  float topd[KNBR];
  int   topi[KNBR];
#pragma unroll
  for (int q = 0; q < KNBR; ++q) { topd[q] = 3.0e38f; topi[q] = 0; }
  float worst = 3.0e38f;
  int   wpos  = 0;

  for (int j0 = 0; j0 < N; j0 += 32) {
    v8f d0 = {};
    d0 = wmma16(a0, load_bt_frag(xb, j0, 0, lane), d0);
    d0 = wmma16(a1, load_bt_frag(xb, j0, 32, lane), d0);
    v8f d1 = {};
    d1 = wmma16(a0, load_bt_frag(xb, j0 + 16, 0, lane), d1);
    d1 = wmma16(a1, load_bt_frag(xb, j0 + 16, 32, lane), d1);

    const int coln = lane & 15;
    const int hi   = lane >> 4;
#pragma unroll
    for (int r = 0; r < 8; ++r) {
      Ds[(r + 8 * hi) * 16 + coln]       = d0[r];
      Ds[256 + (r + 8 * hi) * 16 + coln] = d1[r];
    }
    __syncthreads();

    const float* drow = Ds + tsel * 256 + row * 16;
    const int    jb   = j0 + tsel * 16;
    for (int n = 0; n < 16; ++n) {
      const int   jj  = jb + n;
      const float key = sqb[jj] - 2.f * drow[n];
      if (key < worst) {
#pragma unroll
        for (int q = 0; q < KNBR; ++q)
          if (q == wpos) { topd[q] = key; topi[q] = jj; }
        worst = -3.0e38f;
#pragma unroll
        for (int q = 0; q < KNBR; ++q)
          if (topd[q] > worst) { worst = topd[q]; wpos = q; }
      }
    }
    __syncthreads();
  }

  // merge the two partial lists per row (select 20 smallest of 40)
#pragma unroll
  for (int q = 0; q < KNBR; ++q) {
    Mrg_d[(row * 2 + tsel) * KNBR + q] = topd[q];
    Mrg_i[(row * 2 + tsel) * KNBR + q] = topi[q];
  }
  __syncthreads();
  if (lane < 16) {
    float* md = Mrg_d + lane * 2 * KNBR;
    int*   mi = Mrg_i + lane * 2 * KNBR;
    const size_t obase = ((size_t)b * N + i0 + lane) * KNBR;
    for (int q = 0; q < KNBR; ++q) {
      float best = 3.0e38f;
      int   bp   = 0;
      for (int s = 0; s < 2 * KNBR; ++s)
        if (md[s] < best) { best = md[s]; bp = s; }
      oidx[obase + q] = mi[bp];
      md[bp] = 3.0e38f;
    }
  }
}

// --------------------------------------------- fused EdgeConv (WMMA + LN + max)
// One wave handles 16 points (320 edges = 20 M-tiles of 16).
template <int F, int KP1, int C1, int C2>
__global__ void __launch_bounds__(32)
edgeconv_kernel(const float* __restrict__ xin, const int* __restrict__ nbr,
                const _Float16* __restrict__ W1h, const float* __restrict__ b1,
                const float* __restrict__ g1, const float* __restrict__ be1,
                const _Float16* __restrict__ W2h, const float* __restrict__ b2,
                const float* __restrict__ g2, const float* __restrict__ be2,
                float* __restrict__ yout, int N) {
  __shared__ _Float16 A1[16 * KP1];
  __shared__ float    Hs[16 * C1];
  __shared__ _Float16 A2[16 * C1];
  __shared__ float    Os[16 * C2];
  __shared__ float    Ymax[16 * C2];

  const int lane = threadIdx.x;
  const int nb   = N >> 4;
  const int b    = blockIdx.x / nb;
  const int p0   = (blockIdx.x % nb) << 4;

  for (int t = lane; t < 16 * C2; t += 32) Ymax[t] = -3.0e38f;
  __syncthreads();

  for (int et = 0; et < 20; ++et) {
    {  // build A1 = [x_i | x_j - x_i | 0-pad], lane = (row, half-of-K-range)
      const int row = lane & 15, half = lane >> 4;
      const int le = et * 16 + row;
      const int p  = le / KNBR, kk = le % KNBR;
      const int i  = p0 + p;
      const int j  = nbr[((size_t)b * N + i) * KNBR + kk];
      const float* xi = xin + ((size_t)b * N + i) * F;
      const float* xj = xin + ((size_t)b * N + j) * F;
      const int H = KP1 / 2;
      for (int c = half * H; c < half * H + H; ++c) {
        float v;
        if (c < F)          v = xi[c];
        else if (c < 2 * F) v = xj[c - F] - xi[c - F];
        else                v = 0.f;
        A1[row * KP1 + c] = (_Float16)v;
      }
    }
    __syncthreads();

    // GEMM1: [16 x KP1] x [KP1 x C1]
    for (int ct = 0; ct < C1 / 16; ++ct) {
      v8f acc = {};
#pragma unroll
      for (int kt = 0; kt < KP1 / 32; ++kt) {
        v16h a  = load_a_frag(A1, KP1, kt * 32, lane);
        v16h bf = load_b_frag(W1h, C1, ct * 16, kt * 32, lane);
        acc = wmma16(a, bf, acc);
      }
      const int col = ct * 16 + (lane & 15);
      const float bv = b1[col];
#pragma unroll
      for (int r = 0; r < 8; ++r) {
        const int m = r + (lane >> 4) * 8;
        Hs[m * C1 + col] = acc[r] + bv;
      }
    }
    __syncthreads();
    if (lane < 16) ln_row<_Float16>(Hs + lane * C1, A2 + lane * C1, g1, be1, C1, true);
    __syncthreads();

    // GEMM2: [16 x C1] x [C1 x C2]
    for (int ct = 0; ct < C2 / 16; ++ct) {
      v8f acc = {};
#pragma unroll
      for (int kt = 0; kt < C1 / 32; ++kt) {
        v16h a  = load_a_frag(A2, C1, kt * 32, lane);
        v16h bf = load_b_frag(W2h, C2, ct * 16, kt * 32, lane);
        acc = wmma16(a, bf, acc);
      }
      const int col = ct * 16 + (lane & 15);
      const float bv = b2[col];
#pragma unroll
      for (int r = 0; r < 8; ++r) {
        const int m = r + (lane >> 4) * 8;
        Os[m * C2 + col] = acc[r] + bv;
      }
    }
    __syncthreads();
    if (lane < 16) ln_row<float>(Os + lane * C2, Os + lane * C2, g2, be2, C2, true);
    __syncthreads();

    // max-aggregate edges into their point (rows sequential; lanes over cols)
    for (int r = 0; r < 16; ++r) {
      const int p = (et * 16 + r) / KNBR;
      for (int c = lane; c < C2; c += 32)
        Ymax[p * C2 + c] = fmaxf(Ymax[p * C2 + c], Os[r * C2 + c]);
    }
    __syncthreads();
  }

  for (int t = lane; t < 16 * C2; t += 32) {
    const int p = t / C2, c = t - p * C2;
    yout[((size_t)b * N + p0 + p) * C2 + c] = Ymax[t];
  }
}

// --------------------------------------------------------- global max pooling
__global__ void __launch_bounds__(256)
poolmax_kernel(const float* __restrict__ x1, const float* __restrict__ x2,
               const float* __restrict__ x3, float* __restrict__ pooled, int N) {
  const int b = blockIdx.x, c = threadIdx.x;
  float m = -3.0e38f;
  for (int n = 0; n < N; ++n) {
    float v;
    if (c < 64)       v = x1[((size_t)b * N + n) * 64 + c];
    else if (c < 128) v = x2[((size_t)b * N + n) * 64 + (c - 64)];
    else              v = x3[((size_t)b * N + n) * 128 + (c - 128)];
    m = fmaxf(m, v);
  }
  pooled[b * 256 + c] = m;
}

// -------------------------------------------- global MLP (B=4 rows; tiny, VALU)
__global__ void __launch_bounds__(256)
gm_kernel(const float* __restrict__ pooled, const float* __restrict__ w1,
          const float* __restrict__ b1, const float* __restrict__ g1,
          const float* __restrict__ be1, const float* __restrict__ w2,
          const float* __restrict__ b2, const float* __restrict__ g2,
          const float* __restrict__ be2, float* __restrict__ gout) {
  __shared__ float pin[256];
  __shared__ float h1[512];
  __shared__ float red[256];
  const int b = blockIdx.x, t = threadIdx.x;
  pin[t] = pooled[b * 256 + t];
  __syncthreads();

  float l1[2];
#pragma unroll
  for (int c = 0; c < 2; ++c) {
    const int col = t * 2 + c;
    float s = b1[col];
    for (int k = 0; k < 256; ++k) s += pin[k] * w1[(size_t)k * 512 + col];
    l1[c] = s;
  }
  red[t] = l1[0] + l1[1];
  __syncthreads();
  for (int off = 128; off; off >>= 1) { if (t < off) red[t] += red[t + off]; __syncthreads(); }
  const float m1 = red[0] / 512.f;
  __syncthreads();
  red[t] = (l1[0] - m1) * (l1[0] - m1) + (l1[1] - m1) * (l1[1] - m1);
  __syncthreads();
  for (int off = 128; off; off >>= 1) { if (t < off) red[t] += red[t + off]; __syncthreads(); }
  const float rinv1 = rsqrtf(red[0] / 512.f + LN_EPS);
  __syncthreads();
#pragma unroll
  for (int c = 0; c < 2; ++c) {
    const int col = t * 2 + c;
    h1[col] = fmaxf(g1[col] * (l1[c] - m1) * rinv1 + be1[col], 0.f);
  }
  __syncthreads();

  float l2[4];
#pragma unroll
  for (int c = 0; c < 4; ++c) {
    const int col = t * 4 + c;
    float s = b2[col];
    for (int k = 0; k < 512; ++k) s += h1[k] * w2[(size_t)k * 1024 + col];
    l2[c] = s;
  }
  red[t] = l2[0] + l2[1] + l2[2] + l2[3];
  __syncthreads();
  for (int off = 128; off; off >>= 1) { if (t < off) red[t] += red[t + off]; __syncthreads(); }
  const float m2 = red[0] / 1024.f;
  __syncthreads();
  float vs = 0.f;
#pragma unroll
  for (int c = 0; c < 4; ++c) { float d = l2[c] - m2; vs += d * d; }
  red[t] = vs;
  __syncthreads();
  for (int off = 128; off; off >>= 1) { if (t < off) red[t] += red[t + off]; __syncthreads(); }
  const float rinv2 = rsqrtf(red[0] / 1024.f + LN_EPS);
  __syncthreads();
#pragma unroll
  for (int c = 0; c < 4; ++c) {
    const int col = t * 4 + c;
    gout[b * 1024 + col] = fmaxf(g2[col] * (l2[c] - m2) * rinv2 + be2[col], 0.f);
  }
}

// -------------------------------- per-point head MLP: WMMA GEMM + bias + LN(+ReLU)
// AMODE 0: A from f16 buffer Af16[M][KD];  AMODE 1: A = comb(x1|x2|x3|g) on the fly.
template <int KD, int NOUT, bool RELU, int AMODE, int CTG>
__global__ void __launch_bounds__(32)
rowmlp_kernel(const _Float16* __restrict__ Af16, const float* __restrict__ x1,
              const float* __restrict__ x2, const float* __restrict__ x3,
              const float* __restrict__ gvec, const _Float16* __restrict__ W,
              const float* __restrict__ bias, const float* __restrict__ gg,
              const float* __restrict__ bb, _Float16* __restrict__ outh,
              float* __restrict__ outf) {
  __shared__ _Float16 As[16 * 128];
  __shared__ float    Cs[16 * NOUT];
  const int lane = threadIdx.x;
  const int row0 = blockIdx.x * 16;
  const v8f zacc = {};

  for (int ctg = 0; ctg < NOUT / (16 * CTG); ++ctg) {
    v8f acc[CTG];
#pragma unroll
    for (int c = 0; c < CTG; ++c) acc[c] = zacc;

    for (int kt = 0; kt < KD / 128; ++kt) {
      // stage 16x128 A chunk into LDS (f16)
      for (int t = lane; t < 16 * 128; t += 32) {
        const int r = t >> 7, c = t & 127;
        const int k = kt * 128 + c;
        const size_t gr = (size_t)(row0 + r);
        float v;
        if (AMODE == 0) {
          v = (float)Af16[gr * KD + k];
        } else {
          if (k < 64)       v = x1[gr * 64 + k];
          else if (k < 128) v = x2[gr * 64 + (k - 64)];
          else if (k < 256) v = x3[gr * 128 + (k - 128)];
          else              v = gvec[(gr >> 12) * 1024 + (k - 256)];
        }
        As[r * 128 + c] = (_Float16)v;
      }
      __syncthreads();
      __builtin_prefetch(&W[(size_t)(kt * 128) * NOUT], 0, 1);
#pragma unroll
      for (int ks = 0; ks < 4; ++ks) {
        v16h a = load_a_frag(As, 128, ks * 32, lane);
#pragma unroll
        for (int c = 0; c < CTG; ++c) {
          v16h bf = load_b_frag(W, NOUT, (ctg * CTG + c) * 16, kt * 128 + ks * 32, lane);
          acc[c] = wmma16(a, bf, acc[c]);
        }
      }
      __syncthreads();
    }
#pragma unroll
    for (int c = 0; c < CTG; ++c) {
      const int col = (ctg * CTG + c) * 16 + (lane & 15);
      const float bv = bias[col];
#pragma unroll
      for (int r = 0; r < 8; ++r) {
        const int m = r + (lane >> 4) * 8;
        Cs[m * NOUT + col] = acc[c][r] + bv;
      }
    }
  }
  __syncthreads();

  if (lane < 16) {
    const float* src = Cs + lane * NOUT;
    float m = 0.f;
    for (int c = 0; c < NOUT; ++c) m += src[c];
    m /= (float)NOUT;
    float v = 0.f;
    for (int c = 0; c < NOUT; ++c) { float d = src[c] - m; v += d * d; }
    v /= (float)NOUT;
    const float rinv = rsqrtf(v + LN_EPS);
    const size_t gr = (size_t)(row0 + lane);
    for (int c = 0; c < NOUT; ++c) {
      float t = gg[c] * (src[c] - m) * rinv + bb[c];
      if (RELU) t = fmaxf(t, 0.f);
      if (outh) outh[gr * NOUT + c] = (_Float16)t;
      if (outf) outf[gr * NOUT + c] = t;
    }
  }
}

// ------------------------------------------------------------------ ArcFace
__global__ void __launch_bounds__(256)
arcface_kernel(const float* __restrict__ emb, const float* __restrict__ arcw,
               const int* __restrict__ y, float* __restrict__ out, int BN) {
  const int r = blockIdx.x * 256 + threadIdx.x;
  if (r >= BN) return;
  const float* e = emb + (size_t)r * 128;
  float nrm = 0.f;
  for (int c = 0; c < 128; ++c) nrm += e[c] * e[c];
  nrm = fmaxf(sqrtf(nrm), 1e-12f);
  const int label = y[r] - 1;
  for (int cls = 0; cls < 3; ++cls) {
    const float* w = arcw + cls * 128;
    float wn = 0.f, dot = 0.f;
    for (int c = 0; c < 128; ++c) { wn += w[c] * w[c]; dot += e[c] * w[c]; }
    wn = fmaxf(sqrtf(wn), 1e-12f);
    float cosv = dot / (nrm * wn);
    cosv = fminf(fmaxf(cosv, -1.f), 1.f);
    const float th = acosf(cosv);
    const float mm = (cls == label) ? ARC_M : 0.f;
    out[(size_t)r * 3 + cls] = cosf(th + mm) * ARC_S;
  }
}

// ===================================================================== host
extern "C" void kernel_launch(void* const* d_in, const int* in_sizes, int n_in,
                              void* d_out, int out_size, void* d_ws,
                              size_t ws_size, hipStream_t stream) {
  (void)n_in; (void)out_size; (void)ws_size;
  const int B = 4, N = 4096, BN = B * N;

  enum {
    S_X, S_BATCH, S_Y,
    S_C1W1, S_C1B1, S_C1G1, S_C1BE1, S_C1W2, S_C1B2, S_C1G2, S_C1BE2,
    S_C2W1, S_C2B1, S_C2G1, S_C2BE1, S_C2W2, S_C2B2, S_C2G2, S_C2BE2,
    S_C3W1, S_C3B1, S_C3G1, S_C3BE1, S_C3W2, S_C3B2, S_C3G2, S_C3BE2,
    S_GMW1, S_GMB1, S_GMG1, S_GMBE1, S_GMW2, S_GMB2, S_GMG2, S_GMBE2,
    S_EHW1, S_EHB1, S_EHG1, S_EHBE1, S_EHW2, S_EHB2, S_EHG2, S_EHBE2,
    S_EHW3, S_EHB3, S_EHG3, S_EHBE3,
    S_ARCW, S_NSLOT
  };
  // Order A: pure insertion order (x, batch, y, c1..eh leaves in insertion, arc_w)
  static const int ordA[S_NSLOT] = {
      0, 1, 2, 3, 4, 5, 6, 7, 8, 9, 10, 11, 12, 13, 14, 15, 16,
      17, 18, 19, 20, 21, 22, 23, 24, 25, 26, 27, 28, 29, 30, 31, 32,
      33, 34, 35, 36, 37, 38, 39, 40, 41, 42, 43, 44, 45, 46, 47};
  // Order C: top-level insertion, params sorted (arc_w first, sorted leaves)
  static const int ordC[S_NSLOT] = {
      0, 1, 2,
      10, 4, 8, 6, 11, 5, 9, 7,
      18, 12, 16, 14, 19, 13, 17, 15,
      26, 20, 24, 22, 27, 21, 25, 23,
      46, 40, 44, 42, 47, 41, 45, 43,
      37, 28, 34, 31, 38, 29, 35, 32, 39, 30, 36, 33,
      3};
  // Order B: fully sorted pytree (batch, params{arc_w,c1..c3,eh,gm}, x, y)
  static const int ordB[S_NSLOT] = {
      46, 0, 47,
      8, 2, 6, 4, 9, 3, 7, 5,
      16, 10, 14, 12, 17, 11, 15, 13,
      24, 18, 22, 20, 25, 19, 23, 21,
      44, 38, 42, 40, 45, 39, 43, 41,
      35, 26, 32, 29, 36, 27, 33, 30, 37, 28, 34, 31,
      1};

  const int* ord = ordA;
  if (in_sizes[0] == 98304) ord = (in_sizes[3] == 384) ? ordC : ordA;
  else ord = ordB;

  auto FP = [&](int slot) -> const float* { return (const float*)d_in[ord[slot]]; };
  const float* xptr = FP(S_X);
  const int*   yptr = (const int*)d_in[ord[S_Y]];

  // ---------- workspace carve-out ----------
  char*  wsp = (char*)d_ws;
  size_t off = 0;
  auto take = [&](size_t bytes) -> void* {
    void* p = wsp + off;
    off += (bytes + 255) & ~(size_t)255;
    return p;
  };
  int*      idxb   = (int*)take((size_t)BN * KNBR * 4);
  float*    x1     = (float*)take((size_t)BN * 64 * 4);
  float*    x2     = (float*)take((size_t)BN * 64 * 4);
  float*    x3     = (float*)take((size_t)BN * 128 * 4);
  float*    pooled = (float*)take((size_t)B * 256 * 4);
  float*    gbuf   = (float*)take((size_t)B * 1024 * 4);
  _Float16* h1h    = (_Float16*)take((size_t)BN * 512 * 2);
  _Float16* h2h    = (_Float16*)take((size_t)BN * 256 * 2);
  float*    embf   = (float*)take((size_t)BN * 128 * 4);
  _Float16* xh     = (_Float16*)take((size_t)BN * 64 * 2);  // f16 acts for kNN
  float*    sqb    = (float*)take((size_t)BN * 4);          // ||x||^2 for kNN
  _Float16* c1w1h  = (_Float16*)take(32 * 64 * 2);
  _Float16* c1w2h  = (_Float16*)take(64 * 64 * 2);
  _Float16* c2w1h  = (_Float16*)take(128 * 64 * 2);
  _Float16* c2w2h  = (_Float16*)take(64 * 64 * 2);
  _Float16* c3w1h  = (_Float16*)take(128 * 128 * 2);
  _Float16* c3w2h  = (_Float16*)take(128 * 128 * 2);
  _Float16* ehw1h  = (_Float16*)take((size_t)1280 * 512 * 2);
  _Float16* ehw2h  = (_Float16*)take((size_t)512 * 256 * 2);
  _Float16* ehw3h  = (_Float16*)take((size_t)256 * 128 * 2);

  auto wc = [&](const float* s, _Float16* d, int K, int Nn, int Kpad) {
    int tot = Kpad * Nn;
    wconv_kernel<<<(tot + 255) / 256, 256, 0, stream>>>(s, d, K, Nn, Kpad);
  };
  wc(FP(S_C1W1), c1w1h, 12, 64, 32);
  wc(FP(S_C1W2), c1w2h, 64, 64, 64);
  wc(FP(S_C2W1), c2w1h, 128, 64, 128);
  wc(FP(S_C2W2), c2w2h, 64, 64, 64);
  wc(FP(S_C3W1), c3w1h, 128, 128, 128);
  wc(FP(S_C3W2), c3w2h, 128, 128, 128);
  wc(FP(S_EHW1), ehw1h, 1280, 512, 1280);
  wc(FP(S_EHW2), ehw2h, 512, 256, 512);
  wc(FP(S_EHW3), ehw3h, 256, 128, 256);

  const int knn_grid  = B * (N / 256);
  const int knnw_grid = B * (N / 16);
  const int ec_grid   = B * (N / 16);

  // EdgeConv 1: scalar kNN on 6-dim raw input (too narrow for WMMA)
  knn_kernel<6><<<knn_grid, 256, 0, stream>>>(xptr, idxb, N);
  edgeconv_kernel<6, 32, 64, 64><<<ec_grid, 32, 0, stream>>>(
      xptr, idxb, c1w1h, FP(S_C1B1), FP(S_C1G1), FP(S_C1BE1), c1w2h,
      FP(S_C1B2), FP(S_C1G2), FP(S_C1BE2), x1, N);

  // EdgeConv 2: WMMA Gram kNN on x1
  wc(x1, xh, BN, 64, BN);
  sqnorm_kernel<<<(BN + 255) / 256, 256, 0, stream>>>(x1, sqb, BN);
  knn_wmma_kernel<<<knnw_grid, 32, 0, stream>>>(xh, sqb, idxb, N);
  edgeconv_kernel<64, 128, 64, 64><<<ec_grid, 32, 0, stream>>>(
      x1, idxb, c2w1h, FP(S_C2B1), FP(S_C2G1), FP(S_C2BE1), c2w2h,
      FP(S_C2B2), FP(S_C2G2), FP(S_C2BE2), x2, N);

  // EdgeConv 3: WMMA Gram kNN on x2
  wc(x2, xh, BN, 64, BN);
  sqnorm_kernel<<<(BN + 255) / 256, 256, 0, stream>>>(x2, sqb, BN);
  knn_wmma_kernel<<<knnw_grid, 32, 0, stream>>>(xh, sqb, idxb, N);
  edgeconv_kernel<64, 128, 128, 128><<<ec_grid, 32, 0, stream>>>(
      x2, idxb, c3w1h, FP(S_C3B1), FP(S_C3G1), FP(S_C3BE1), c3w2h,
      FP(S_C3B2), FP(S_C3G2), FP(S_C3BE2), x3, N);

  // Global pooling + global MLP
  poolmax_kernel<<<B, 256, 0, stream>>>(x1, x2, x3, pooled, N);
  gm_kernel<<<B, 256, 0, stream>>>(pooled, FP(S_GMW1), FP(S_GMB1), FP(S_GMG1),
                                   FP(S_GMBE1), FP(S_GMW2), FP(S_GMB2),
                                   FP(S_GMG2), FP(S_GMBE2), gbuf);

  // Head MLP (comb = [x1|x2|x3|g] built on the fly in layer 1)
  const int rm_grid = BN / 16;
  rowmlp_kernel<1280, 512, true, 1, 4><<<rm_grid, 32, 0, stream>>>(
      nullptr, x1, x2, x3, gbuf, ehw1h, FP(S_EHB1), FP(S_EHG1), FP(S_EHBE1),
      h1h, nullptr);
  rowmlp_kernel<512, 256, true, 0, 4><<<rm_grid, 32, 0, stream>>>(
      h1h, nullptr, nullptr, nullptr, nullptr, ehw2h, FP(S_EHB2), FP(S_EHG2),
      FP(S_EHBE2), h2h, nullptr);
  rowmlp_kernel<256, 128, false, 0, 2><<<rm_grid, 32, 0, stream>>>(
      h2h, nullptr, nullptr, nullptr, nullptr, ehw3h, FP(S_EHB3), FP(S_EHG3),
      FP(S_EHBE3), nullptr, embf);

  arcface_kernel<<<(BN + 255) / 256, 256, 0, stream>>>(
      embf, FP(S_ARCW), yptr, (float*)d_out, BN);
}
